// ModelA_GNN_A100_58007828300476
// MI455X (gfx1250) — matrix operand order repaired and checked by reading, not compiled, not run on hardware
//
#include <hip/hip_runtime.h>

#define BB 2
#define NN 512
#define DD 64
#define BNN (BB * NN * NN)
#define EPS_BN 1e-5f

typedef __attribute__((ext_vector_type(16))) __bf16 v16bf;
typedef __attribute__((ext_vector_type(8)))  float  v8f;

__device__ __forceinline__ unsigned short f2bf_bits(float f) {
  __bf16 h = (__bf16)f;
  return __builtin_bit_cast(unsigned short, h);
}

// ---------------------------------------------------------------------------
// Setup 1: fold BN1 and split W1 across the concat:
//   A'[b,n,o] = s1[o]*(f[b,n]·W1[o,0:64] + b1[o]) + be1[o] - m1[o]*s1[o]
//   B'[b,n,o] = s1[o]*(f[b,n]·W1[o,64:128])
// ---------------------------------------------------------------------------
__global__ void ep_precomp_ab(const float* __restrict__ feat,
                              const float* __restrict__ W1, const float* __restrict__ b1,
                              const float* __restrict__ g1, const float* __restrict__ be1,
                              const float* __restrict__ m1, const float* __restrict__ v1,
                              float* __restrict__ Ap, float* __restrict__ Bp) {
  __shared__ float fs[DD];
  int bn = blockIdx.x;
  int o  = threadIdx.x;               // 0..127
  if (o < DD) fs[o] = feat[bn * DD + o];
  __syncthreads();
  const float* w = W1 + o * (2 * DD);
  float d0 = 0.f, d1 = 0.f;
#pragma unroll
  for (int k = 0; k < DD; ++k) { d0 += fs[k] * w[k]; d1 += fs[k] * w[DD + k]; }
  float s = g1[o] * rsqrtf(v1[o] + EPS_BN);
  Ap[bn * 128 + o] = s * d0 + s * b1[o] + be1[o] - m1[o] * s;
  Bp[bn * 128 + o] = s * d1;
}

// ---------------------------------------------------------------------------
// Setup 2: fold BN2 into W2 and pre-pack into WMMA B-fragment lane order.
// Fragment f = c*4+n (K-chunk c of 32, N-tile n of 16). For lane l, elem e:
//   N = n*16 + (l%16),  K = c*32 + (l/16)*16 + e      (contiguous k order)
// ---------------------------------------------------------------------------
__global__ void ep_pack_w2(const float* __restrict__ W2, const float* __restrict__ b2,
                           const float* __restrict__ g2, const float* __restrict__ be2,
                           const float* __restrict__ m2, const float* __restrict__ v2,
                           unsigned short* __restrict__ w2f, float* __restrict__ b2fold) {
  int idx = blockIdx.x * blockDim.x + threadIdx.x;
  if (idx < 4 * 4 * 32 * 16) {
    int e    = idx & 15;
    int lane = (idx >> 4) & 31;
    int f    = idx >> 9;              // 0..15
    int n = f & 3, c = f >> 2;
    int o = n * 16 + (lane & 15);
    int k = c * 32 + (lane >> 4) * 16 + e;
    float s = g2[o] * rsqrtf(v2[o] + EPS_BN);
    w2f[idx] = f2bf_bits(s * W2[o * 128 + k]);
  }
  if (idx < 64) {
    float s = g2[idx] * rsqrtf(v2[idx] + EPS_BN);
    b2fold[idx] = s * b2[idx] + be2[idx] - m2[idx] * s;
  }
}

__global__ void ep_pack_w3(const float* __restrict__ W3, const float* __restrict__ b3,
                           const float* __restrict__ g3, const float* __restrict__ be3,
                           const float* __restrict__ m3, const float* __restrict__ v3,
                           unsigned short* __restrict__ w3f, float* __restrict__ b3fold) {
  int idx = blockIdx.x * blockDim.x + threadIdx.x;
  if (idx < 2 * 2 * 32 * 16) {
    int e    = idx & 15;
    int lane = (idx >> 4) & 31;
    int f    = idx >> 9;              // 0..3
    int n = f & 1, c = f >> 1;
    int o = n * 16 + (lane & 15);
    int k = c * 32 + (lane >> 4) * 16 + e;
    float s = g3[o] * rsqrtf(v3[o] + EPS_BN);
    w3f[idx] = f2bf_bits(s * W3[o * 64 + k]);
  }
  if (idx < 32) {
    float s = g3[idx] * rsqrtf(v3[idx] + EPS_BN);
    b3fold[idx] = s * b3[idx] + be3[idx] - m3[idx] * s;
  }
}

// ---------------------------------------------------------------------------
// Setup 3: normalized features for cosine similarity; valid-node counts.
// ---------------------------------------------------------------------------
__global__ void ep_normalize(const float* __restrict__ feat, float* __restrict__ nf) {
  __shared__ float fs[DD];
  int bn = blockIdx.x, d = threadIdx.x;   // 64 threads
  fs[d] = feat[bn * DD + d];
  __syncthreads();
  float ss = 0.f;
#pragma unroll
  for (int k = 0; k < DD; ++k) ss += fs[k] * fs[k];
  float nrm = fmaxf(sqrtf(ss), 1e-12f);
  nf[bn * DD + d] = fs[d] / nrm;
}

__global__ void ep_nvalid(const int* __restrict__ masks, int* __restrict__ nvalid) {
  int b = threadIdx.x;
  if (b < BB) {
    int s = 0;
    for (int j = 0; j < NN; ++j) s += (masks[b * NN + j] != 0);
    nvalid[b] = s;
  }
}

// ---------------------------------------------------------------------------
// Hot kernel: one wave32 per tile of 16 pairs (fixed b,i; j = j0..j0+15).
//   h1 = relu(A'[i] + B'[j])                (built directly as A-fragments)
//   h2 = relu(h1 @ W2f^T + b2f)             16x v_wmma_f32_16x16x32_bf16
//   h3 = relu(h2 @ W3f^T + b3f)              4x v_wmma_f32_16x16x32_bf16
//   raw = h3 @ W4^T + b4                    shfl-xor reduction
// A-fragment layout (16-bit A 16x32, contiguous k order):
//   lane l: M = l%16 ; elem e -> K = chunk*32 + (l/16)*16 + e
// ---------------------------------------------------------------------------
__global__ void __launch_bounds__(32)
ep_mlp_wmma(const float* __restrict__ Ap, const float* __restrict__ Bp,
            const unsigned short* __restrict__ w2f, const float* __restrict__ b2f,
            const unsigned short* __restrict__ w3f, const float* __restrict__ b3f,
            const float* __restrict__ W4, const float* __restrict__ b4,
            float* __restrict__ raw) {
  __shared__ __align__(16) float Ash[128];
  __shared__ __align__(16) float Bsh[16 * 128];   // later reused as h2sh[16][64]

  int t   = blockIdx.x;
  int b   = t / (NN * (NN / 16));
  int rem = t % (NN * (NN / 16));
  int i   = rem / (NN / 16);
  int j0  = (rem % (NN / 16)) * 16;
  int l   = threadIdx.x;
  int m   = l & 15;                 // M index / N index inside a tile
  int h   = l >> 4;                 // half-wave select

  // Load A'[b,i,:] (128 f32) and B'[b,j0..j0+15,:] (16x128 f32) into LDS.
#pragma unroll
  for (int q = 0; q < 4; ++q) Ash[q * 32 + l] = Ap[(b * NN + i) * 128 + q * 32 + l];
  const float4* Bp4  = (const float4*)(Bp + (size_t)(b * NN + j0) * 128);
  float4*       Bsh4 = (float4*)Bsh;
#pragma unroll
  for (int q = 0; q < 16; ++q) Bsh4[q * 32 + l] = Bp4[q * 32 + l];
  __syncthreads();

  // Build layer-2 A-fragments: h1 = relu(A' + B'), converted to bf16.
  v16bf a2[4];
#pragma unroll
  for (int c = 0; c < 4; ++c) {
#pragma unroll
    for (int e = 0; e < 16; ++e) {
      int k = c * 32 + h * 16 + e;
      float val = Ash[k] + Bsh[m * 128 + k];
      a2[c][e] = (__bf16)fmaxf(val, 0.f);
    }
  }

  // Layer 2: [16 pairs x 128] @ [128 x 64] via 16 WMMAs.
  v8f acc2[4] = {};
  const v16bf* w2v = (const v16bf*)w2f;
#pragma unroll
  for (int c = 0; c < 4; ++c) {
#pragma unroll
    for (int n = 0; n < 4; ++n) {
      acc2[n] = __builtin_amdgcn_wmma_f32_16x16x32_bf16(
          false, a2[c], false, w2v[(c * 4 + n) * 32 + l], (short)0, acc2[n],
          false, false);
    }
  }

  // h2 = relu(acc2 + b2f); transpose via LDS (D layout: lane=N, reg r -> M=r+8h).
  __syncthreads();
  float* h2sh = Bsh;                // reuse: 16 rows x 64 cols
#pragma unroll
  for (int n = 0; n < 4; ++n) {
    int o = n * 16 + m;
    float bb = b2f[o];
#pragma unroll
    for (int r = 0; r < 8; ++r) {
      int row = r + h * 8;
      h2sh[row * 64 + o] = fmaxf(acc2[n][r] + bb, 0.f);
    }
  }
  __syncthreads();

  // Layer 3 A-fragments from h2sh.
  v16bf a3[2];
#pragma unroll
  for (int c = 0; c < 2; ++c) {
#pragma unroll
    for (int e = 0; e < 16; ++e) {
      int k = c * 32 + h * 16 + e;
      a3[c][e] = (__bf16)h2sh[m * 64 + k];
    }
  }

  // Layer 3: [16 x 64] @ [64 x 32] via 4 WMMAs.
  v8f acc3[2] = {};
  const v16bf* w3v = (const v16bf*)w3f;
#pragma unroll
  for (int c = 0; c < 2; ++c) {
#pragma unroll
    for (int n = 0; n < 2; ++n) {
      acc3[n] = __builtin_amdgcn_wmma_f32_16x16x32_bf16(
          false, a3[c], false, w3v[(c * 2 + n) * 32 + l], (short)0, acc3[n],
          false, false);
    }
  }

  // Layer 4: scalar head. Lane covers channels m and m+16 over rows r+8h.
  float w4v0 = W4[m],        w4v1 = W4[16 + m];
  float b3v0 = b3f[m],       b3v1 = b3f[16 + m];
  float part[8];
#pragma unroll
  for (int r = 0; r < 8; ++r) {
    part[r] = fmaxf(acc3[0][r] + b3v0, 0.f) * w4v0 +
              fmaxf(acc3[1][r] + b3v1, 0.f) * w4v1;
  }
#pragma unroll
  for (int r = 0; r < 8; ++r) {
    float v = part[r];
    v += __shfl_xor(v, 8, 32);
    v += __shfl_xor(v, 4, 32);
    v += __shfl_xor(v, 2, 32);
    v += __shfl_xor(v, 1, 32);
    part[r] = v;
  }
  if (m == 0) {
    float bb4 = b4[0];
#pragma unroll
    for (int r = 0; r < 8; ++r) {
      raw[(size_t)(b * NN + i) * NN + j0 + h * 8 + r] = part[r] + bb4;
    }
  }
}

// ---------------------------------------------------------------------------
// Pass 2: diag -> -10, symmetrize, cosine boost, sigmoid/threshold/clip, mask.
// ---------------------------------------------------------------------------
__global__ void ep_finalize(const float* __restrict__ raw, const float* __restrict__ nf,
                            const int* __restrict__ masks, const int* __restrict__ nvalid,
                            float* __restrict__ out) {
  int idx = blockIdx.x * blockDim.x + threadIdx.x;
  if (idx >= BNN) return;
  int b = idx / (NN * NN);
  int r = idx % (NN * NN);
  int i = r / NN, j = r % NN;
  float lv  = (i == j) ? -10.f : raw[idx];
  float lvt = (i == j) ? -10.f : raw[(size_t)b * NN * NN + (size_t)j * NN + i];
  float sym = 0.5f * (lv + lvt);
  const float4* ni = (const float4*)(nf + (size_t)(b * NN + i) * DD);
  const float4* nj = (const float4*)(nf + (size_t)(b * NN + j) * DD);
  float sim = 0.f;
#pragma unroll
  for (int q = 0; q < DD / 4; ++q) {
    float4 a = ni[q], c = nj[q];
    sim += a.x * c.x + a.y * c.y + a.z * c.z + a.w * c.w;
  }
  float logit = (sym + 2.f * sim) * 2.f;          // temperature 0.5
  float p = 1.f / (1.f + __expf(-logit));
  p = (p > 0.6f) ? p * 1.2f : p * 0.8f;
  p = fminf(fmaxf(p, 0.f), 1.f);
  bool ok = (masks[b * NN + i] != 0) && (masks[b * NN + j] != 0) && (nvalid[b] > 1);
  out[idx]        = ok ? p : 0.f;
  out[BNN + idx]  = ok ? logit : 0.f;
}

// ---------------------------------------------------------------------------
extern "C" void kernel_launch(void* const* d_in, const int* in_sizes, int n_in,
                              void* d_out, int out_size, void* d_ws, size_t ws_size,
                              hipStream_t stream) {
  (void)in_sizes; (void)n_in; (void)out_size; (void)ws_size;
  const float* feat  = (const float*)d_in[0];
  const int*   masks = (const int*)d_in[1];
  const float* W1 = (const float*)d_in[2];  const float* b1 = (const float*)d_in[3];
  const float* W2 = (const float*)d_in[4];  const float* b2 = (const float*)d_in[5];
  const float* W3 = (const float*)d_in[6];  const float* b3 = (const float*)d_in[7];
  const float* W4 = (const float*)d_in[8];  const float* b4 = (const float*)d_in[9];
  const float* g1 = (const float*)d_in[10]; const float* be1 = (const float*)d_in[11];
  const float* m1 = (const float*)d_in[12]; const float* v1  = (const float*)d_in[13];
  const float* g2 = (const float*)d_in[14]; const float* be2 = (const float*)d_in[15];
  const float* m2 = (const float*)d_in[16]; const float* v2  = (const float*)d_in[17];
  const float* g3 = (const float*)d_in[18]; const float* be3 = (const float*)d_in[19];
  const float* m3 = (const float*)d_in[20]; const float* v3  = (const float*)d_in[21];

  char* ws = (char*)d_ws;
  size_t off = 0;
  auto carve = [&](size_t bytes) -> void* {
    void* p = ws + off;
    off += (bytes + 255) & ~(size_t)255;
    return p;
  };
  float*          Ap     = (float*)carve((size_t)BB * NN * 128 * sizeof(float));
  float*          Bp     = (float*)carve((size_t)BB * NN * 128 * sizeof(float));
  float*          nf     = (float*)carve((size_t)BB * NN * DD * sizeof(float));
  float*          b2f    = (float*)carve(64 * sizeof(float));
  float*          b3f    = (float*)carve(32 * sizeof(float));
  unsigned short* w2frag = (unsigned short*)carve(8192 * sizeof(unsigned short));
  unsigned short* w3frag = (unsigned short*)carve(2048 * sizeof(unsigned short));
  int*            nvalid = (int*)carve(4 * sizeof(int));
  float*          raw    = (float*)carve((size_t)BNN * sizeof(float));

  ep_precomp_ab<<<BB * NN, 128, 0, stream>>>(feat, W1, b1, g1, be1, m1, v1, Ap, Bp);
  ep_pack_w2<<<32, 256, 0, stream>>>(W2, b2, g2, be2, m2, v2, w2frag, b2f);
  ep_pack_w3<<<8, 256, 0, stream>>>(W3, b3, g3, be3, m3, v3, w3frag, b3f);
  ep_normalize<<<BB * NN, 64, 0, stream>>>(feat, nf);
  ep_nvalid<<<1, 32, 0, stream>>>(masks, nvalid);
  ep_mlp_wmma<<<BB * NN * (NN / 16), 32, 0, stream>>>(Ap, Bp, w2frag, b2f,
                                                      w3frag, b3f, W4, b4, raw);
  ep_finalize<<<(BNN + 255) / 256, 256, 0, stream>>>(raw, nf, masks, nvalid,
                                                     (float*)d_out);
}
